// UnrolledLSTM_27874337751802
// MI455X (gfx1250) — compile-verified, matrix operand order checked by reading, not run on hardware
//
#include <hip/hip_runtime.h>
#include <hip/hip_bf16.h>

typedef _Float16 f16;
typedef _Float16 v16h __attribute__((ext_vector_type(16)));
typedef _Float16 v8h  __attribute__((ext_vector_type(8)));
typedef float    v8f  __attribute__((ext_vector_type(8)));

#define LSTM_BATCH 1024
#define LSTM_SEQ   128
#define LSTM_INS   512
#define LSTM_HID   512
#define LSTM_ZD    1024   // INS + HID
#define KC         32     // K chunk width (one WMMA K)
#define LDS_STRIDE 40     // padded row stride in halfs (80B = 20 banks -> conflict-free)

__device__ __forceinline__ float fast_sigmoid(float x) {
    return 1.0f / (1.0f + __expf(-x));
}
__device__ __forceinline__ float fast_tanh(float x) {
    float e = __expf(2.0f * x);
    return (e - 1.0f) / (e + 1.0f);
}

// Build a 16-half WMMA fragment from two 16B LDS chunks (stays addrspace(3)).
#define LDS_FRAG(buf, row, kh)                                                 \
    __builtin_shufflevector(                                                   \
        *(const v8h*)&(buf)[(row) * LDS_STRIDE + (kh)],                        \
        *(const v8h*)&(buf)[(row) * LDS_STRIDE + (kh) + 16],                   \
        0,1,2,3,4,5,6,7,8,9,10,11,12,13,14,15)

// ---------------------------------------------------------------------------
// One-time: convert the four gate weight matrices (512 x 1024 f32, row-major,
// k contiguous) into f16, packed [gate][n][k] in workspace.
// ---------------------------------------------------------------------------
__global__ void lstm_convert_weights(const float* __restrict__ Wf,
                                     const float* __restrict__ Wu,
                                     const float* __restrict__ Wc,
                                     const float* __restrict__ Wo,
                                     f16* __restrict__ W16) {
    int i = blockIdx.x * blockDim.x + threadIdx.x;   // 0 .. HID*ZD
    const int N = LSTM_HID * LSTM_ZD;
    if (i < N) {
        W16[0 * N + i] = (f16)Wf[i];
        W16[1 * N + i] = (f16)Wu[i];
        W16[2 * N + i] = (f16)Wc[i];
        W16[3 * N + i] = (f16)Wo[i];
    }
}

// ---------------------------------------------------------------------------
// One-time: build z0 = [x[:,0,:], h0=0] in f16, zero the c state.
// ---------------------------------------------------------------------------
__global__ void lstm_init(const float* __restrict__ x,
                          f16* __restrict__ z0,
                          float* __restrict__ cstate) {
    int i = blockIdx.x * blockDim.x + threadIdx.x;   // 0 .. BATCH*ZD
    int m = i >> 10;           // ZD == 1024
    int k = i & 1023;
    if (k < LSTM_INS) {
        z0[i] = (f16)x[(size_t)m * LSTM_SEQ * LSTM_INS + k];   // t = 0
    } else {
        z0[i] = (f16)0.0f;
        cstate[(size_t)m * LSTM_HID + (k - LSTM_INS)] = 0.0f;
    }
}

// ---------------------------------------------------------------------------
// One LSTM timestep, fused GEMM (4 gates) + gates + state update.
// Block = 256 threads = 8 wave32 (4m x 2n). Block tile: 128(M) x 32(N) x 4 gates.
// Double-buffered LDS staging of A (z) and B (W) K-chunks, software-pipelined
// with global prefetch of chunk k+1 while computing chunk k from LDS.
// The K loop is FULLY UNROLLED (32 iterations) so the buffer parity is a
// compile-time constant (immediate LDS offsets), loads clause together, and
// the 8 WMMAs per chunk issue back-to-back behind a single dscnt wait.
// Grid = (BATCH/128, HID/32).
// ---------------------------------------------------------------------------
__global__ __launch_bounds__(256)
void lstm_step(const f16* __restrict__ z,        // [BATCH][ZD]  f16
               f16*       __restrict__ znext,    // [BATCH][ZD]  f16
               const f16* __restrict__ W16,      // [4][HID][ZD] f16
               const float* __restrict__ bf, const float* __restrict__ bu,
               const float* __restrict__ bc, const float* __restrict__ bo,
               float*     __restrict__ cstate,   // [BATCH][HID] f32
               const float* __restrict__ x,      // [BATCH][SEQ][INS] f32
               float*     __restrict__ out,      // d_out base
               int t, int is_last) {
    // A tile: 128 rows x 32 k ; B tile: (4 gates * 32 cols) rows x 32 k
    __shared__ __attribute__((aligned(16))) f16 As[2][128 * LDS_STRIDE];
    __shared__ __attribute__((aligned(16))) f16 Bs[2][128 * LDS_STRIDE];

    const int tid   = threadIdx.x;
    const int lane  = tid & 31;
    const int wave  = tid >> 5;
    const int mwave = wave >> 1;                      // 0..3
    const int nwave = wave & 1;                       // 0..1
    const int m_blk = blockIdx.x * 128;               // batch tile base
    const int n_blk = blockIdx.y * 32;                // hidden tile base

    const int lrow  = lane & 15;
    const int khalf = (lane >> 4) * 8;                // 0 or 8 (A/B lane layout)

    // ---- per-thread staging units: u0 = tid, u1 = tid+256 ------------------
    // unit u: row = u>>2 (0..127), seg = u&3 (16B k-segment within 32-k chunk)
    const int u0 = tid, u1 = tid + 256;
    const int r0 = u0 >> 2, s0 = (u0 & 3) * 8;
    const int r1 = u1 >> 2, s1 = (u1 & 3) * 8;
    const f16* gA0 = z + (size_t)(m_blk + r0) * LSTM_ZD + s0;
    const f16* gA1 = z + (size_t)(m_blk + r1) * LSTM_ZD + s1;
    // B row r (0..127): gate = r>>5, ncol = r&31
    const f16* gB0 = W16 + (size_t)(r0 >> 5) * (LSTM_HID * LSTM_ZD)
                         + (size_t)(n_blk + (r0 & 31)) * LSTM_ZD + s0;
    const f16* gB1 = W16 + (size_t)(r1 >> 5) * (LSTM_HID * LSTM_ZD)
                         + (size_t)(n_blk + (r1 & 31)) * LSTM_ZD + s1;
    const int l0 = r0 * LDS_STRIDE + s0;
    const int l1 = r1 * LDS_STRIDE + s1;

    const v8f vzero = {0.f, 0.f, 0.f, 0.f, 0.f, 0.f, 0.f, 0.f};
    v8f acc[4][2];
#pragma unroll
    for (int g = 0; g < 4; ++g)
#pragma unroll
        for (int s = 0; s < 2; ++s) acc[g][s] = vzero;

    // ---- prologue: stage chunk 0 into buffer 0 -----------------------------
    {
        v8h ra0 = *(const v8h*)(gA0);
        v8h ra1 = *(const v8h*)(gA1);
        v8h rb0 = *(const v8h*)(gB0);
        v8h rb1 = *(const v8h*)(gB1);
        *(v8h*)&As[0][l0] = ra0;
        *(v8h*)&As[0][l1] = ra1;
        *(v8h*)&Bs[0][l0] = rb0;
        *(v8h*)&Bs[0][l1] = rb1;
    }
    __syncthreads();

#pragma unroll
    for (int kcIdx = 0; kcIdx < LSTM_ZD / KC; ++kcIdx) {
        const int p  = kcIdx & 1;        // compile-time after full unroll
        const int nk = (kcIdx + 1) * KC;
        v8h ra0, ra1, rb0, rb1;
        const bool more = (nk < LSTM_ZD);
        if (more) {                       // global prefetch of next chunk
            ra0 = *(const v8h*)(gA0 + nk);
            ra1 = *(const v8h*)(gA1 + nk);
            rb0 = *(const v8h*)(gB0 + nk);
            rb1 = *(const v8h*)(gB1 + nk);
        }

        // ---- load ALL fragments for this chunk first (one dscnt wait) ------
        v16h a0 = LDS_FRAG(As[p], mwave * 32 + lrow,      khalf);
        v16h a1 = LDS_FRAG(As[p], mwave * 32 + 16 + lrow, khalf);
        v16h bfr[4];
#pragma unroll
        for (int g = 0; g < 4; ++g)
            bfr[g] = LDS_FRAG(Bs[p], g * 32 + nwave * 16 + lrow, khalf);

        // ---- WMMA burst: 8 back-to-back matrix ops -------------------------
#pragma unroll
        for (int g = 0; g < 4; ++g) {
            acc[g][0] = __builtin_amdgcn_wmma_f32_16x16x32_f16(
                false, a0, false, bfr[g], (short)0, acc[g][0], false, false);
            acc[g][1] = __builtin_amdgcn_wmma_f32_16x16x32_f16(
                false, a1, false, bfr[g], (short)0, acc[g][1], false, false);
        }

        if (more) {                       // stage next chunk into buffer p^1
            *(v8h*)&As[p ^ 1][l0] = ra0;
            *(v8h*)&As[p ^ 1][l1] = ra1;
            *(v8h*)&Bs[p ^ 1][l0] = rb0;
            *(v8h*)&Bs[p ^ 1][l1] = rb1;
        }
        __syncthreads();
    }

    // ---------------- epilogue: gates + state update -----------------------
    const int n   = n_blk + nwave * 16 + lrow;  // hidden index (D-matrix N=lane)
    const int mhi = (lane >> 4) * 8;            // D rows: lanes 16-31 -> M = 8+r
    const float vbf = bf[n], vbu = bu[n], vbc = bc[n], vbo = bo[n];

#pragma unroll
    for (int s = 0; s < 2; ++s) {
#pragma unroll
        for (int r = 0; r < 8; ++r) {
            const int m = m_blk + mwave * 32 + s * 16 + mhi + r;
            float df = acc[0][s][r] + vbf;
            float du = acc[1][s][r] + vbu;
            float dg = acc[2][s][r] + vbc;
            float dq = acc[3][s][r] + vbo;
            float fg = fast_sigmoid(df);
            float ug = fast_sigmoid(du);
            float gg = fast_tanh(dg);
            float og = fast_sigmoid(dq);

            const size_t ci = (size_t)m * LSTM_HID + n;
            float cnew = fg * cstate[ci] + ug * gg;
            cstate[ci] = cnew;
            float h = og * fast_tanh(cnew);

            out[(size_t)t * LSTM_BATCH * LSTM_HID + ci] = h;          // hs[t]
            znext[(size_t)m * LSTM_ZD + LSTM_HID + n] = (f16)h;       // h half

            if (is_last) {
                out[(size_t)LSTM_SEQ * LSTM_BATCH * LSTM_HID + ci] = h;          // h_T
                out[(size_t)(LSTM_SEQ + 1) * LSTM_BATCH * LSTM_HID + ci] = cnew; // c_T
            } else {
                // stage x[:, t+1, :] into the input half of z_next
                znext[(size_t)m * LSTM_ZD + n] =
                    (f16)x[(size_t)m * LSTM_SEQ * LSTM_INS
                           + (size_t)(t + 1) * LSTM_INS + n];
            }
        }
    }
}

// ---------------------------------------------------------------------------
extern "C" void kernel_launch(void* const* d_in, const int* in_sizes, int n_in,
                              void* d_out, int out_size, void* d_ws, size_t ws_size,
                              hipStream_t stream) {
    const float* x  = (const float*)d_in[0];
    const float* Wf = (const float*)d_in[1];
    const float* bf = (const float*)d_in[2];
    const float* Wu = (const float*)d_in[3];
    const float* bu = (const float*)d_in[4];
    const float* Wc = (const float*)d_in[5];
    const float* bc = (const float*)d_in[6];
    const float* Wo = (const float*)d_in[7];
    const float* bo = (const float*)d_in[8];
    float* out = (float*)d_out;

    // workspace layout:
    //   W16   : 4 * 512 * 1024 f16  = 4 MB
    //   z0,z1 : 2 * 1024 * 1024 f16 = 4 MB
    //   c     : 1024 * 512 f32      = 2 MB
    char* ws = (char*)d_ws;
    f16* W16 = (f16*)ws;
    f16* z0  = (f16*)(ws + (size_t)4 * LSTM_HID * LSTM_ZD * sizeof(f16));
    f16* z1  = z0 + (size_t)LSTM_BATCH * LSTM_ZD;
    float* cstate = (float*)(ws + (size_t)4 * LSTM_HID * LSTM_ZD * sizeof(f16)
                                + (size_t)2 * LSTM_BATCH * LSTM_ZD * sizeof(f16));

    lstm_convert_weights<<<(LSTM_HID * LSTM_ZD) / 256, 256, 0, stream>>>(
        Wf, Wu, Wc, Wo, W16);
    lstm_init<<<(LSTM_BATCH * LSTM_ZD) / 256, 256, 0, stream>>>(x, z0, cstate);

    dim3 grid(LSTM_BATCH / 128, LSTM_HID / 32);
    for (int t = 0; t < LSTM_SEQ; ++t) {
        f16* zc = (t & 1) ? z1 : z0;
        f16* zn = (t & 1) ? z0 : z1;
        lstm_step<<<grid, dim3(256), 0, stream>>>(
            zc, zn, W16, bf, bu, bc, bo, cstate, x, out, t, (t == LSTM_SEQ - 1) ? 1 : 0);
    }
}
